// Qwen3_5VisionAttention_7516192768263
// MI455X (gfx1250) — matrix operand order repaired and checked by reading, not compiled, not run on hardware
//
#include <hip/hip_runtime.h>
#include <hip/hip_bf16.h>
#include <stdint.h>

// ---------------------------------------------------------------------------
// Types for CDNA5 WMMA (wave32): bf16 16x16x32, f32 accumulate.
// ---------------------------------------------------------------------------
typedef __attribute__((ext_vector_type(16))) __bf16        v16bf;
typedef __attribute__((ext_vector_type(8)))  float         v8f;
typedef __attribute__((ext_vector_type(4)))  unsigned int  u32x4;
typedef __attribute__((ext_vector_type(8)))  int           i32x8;
typedef __attribute__((ext_vector_type(4)))  int           i32x4;

union Frag16 {
    u32x4          q[2];
    v16bf          v;
    unsigned short s[16];
};

__device__ __forceinline__ unsigned short f2bf(float f) {
    unsigned int u = __float_as_uint(f);
    u += 0x7FFFu + ((u >> 16) & 1u);   // round-to-nearest-even
    return (unsigned short)(u >> 16);
}

__device__ __forceinline__ v16bf load_frag16(const unsigned short* p0,
                                             const unsigned short* p1) {
    Frag16 f;
    f.q[0] = *(const u32x4*)p0;   // 8 bf16
    f.q[1] = *(const u32x4*)p1;   // 8 bf16
    return f.v;
}

__device__ __forceinline__ v8f v8f_zero() {
    v8f z;
#pragma unroll
    for (int i = 0; i < 8; ++i) z[i] = 0.f;
    return z;
}

// ---------------------------------------------------------------------------
// TDM: async DMA of a 2D bf16 tile (tileX x tileY) from a row-major tensor
// [dimY][dimX] (element stride strideX) into LDS, with per-row LDS padding.
// D# bitfields per CDNA5 ISA ch.8 (group0: count/lds_addr/global_addr/type,
// group1: data_size/pad/tensor dims/tile dims/strides). Groups 2/3 zero
// (tile_dim2/3 = 0 -> unused, 2-D tensor). Tracked by TENSORcnt.
// This toolchain exposes the 6-arg clang-23 builtin form:
//   (u32x4 g0, i32x8 g1, i32x4, i32x4, i32x8, i32 cpol)
// ---------------------------------------------------------------------------
__device__ __forceinline__ void tdm_load_2d_pad_bf16(
    unsigned lds_addr, const unsigned short* gptr,
    int dimX, int dimY, int tileX, int tileY, int strideX,
    int pad_interval_code, int pad_amount_code)
{
    unsigned long long ga = (unsigned long long)(uintptr_t)gptr;  // byte addr
    u32x4 g0;
    g0.x = 1u;                                            // count=1, user mode
    g0.y = lds_addr;                                      // LDS byte address
    g0.z = (unsigned)(ga & 0xFFFFFFFFu);                  // global_addr[31:0]
    g0.w = (unsigned)((ga >> 32) & 0x01FFFFFFu)           // global_addr[56:32]
         | (2u << 30);                                    // type=2 ("image")
    i32x8 g1;
    g1[0] = (1 << 16)                                     // data_size: 2 bytes
          | (1 << 20)                                     // pad_enable
          | (pad_interval_code << 22)                     // DWORDs per row chunk
          | (pad_amount_code << 25);                      // pad DWORDs - 1
    g1[1] = (dimX & 0xFFFF) << 16;                        // tensor_dim0 lo16
    g1[2] = ((unsigned)dimX >> 16) | ((dimY & 0xFFFF) << 16); // dim0 hi / dim1 lo
    g1[3] = ((unsigned)dimY >> 16) | ((tileX & 0xFFFF) << 16);// dim1 hi / tile0
    g1[4] = (tileY & 0xFFFF);                             // tile_dim1, tile_dim2=0
    g1[5] = strideX;                                      // tensor_dim0_stride lo32
    g1[6] = 0;                                            // stride hi / dim1_stride lo
    g1[7] = 0;
    i32x4 z4 = {0, 0, 0, 0};
    i32x8 z8 = {0, 0, 0, 0, 0, 0, 0, 0};
    __builtin_amdgcn_tensor_load_to_lds(g0, g1, z4, z4, z8, 0);
}

// ---------------------------------------------------------------------------
// Kernel: f32 -> bf16 convert
// ---------------------------------------------------------------------------
__global__ void cvt_f32_bf16(const float* __restrict__ src,
                             unsigned short* __restrict__ dst, int n) {
    int i = blockIdx.x * 256 + threadIdx.x;
    if (i < n) dst[i] = f2bf(src[i]);
}

// ---------------------------------------------------------------------------
// Kernel: C[M,N] = bf16 A[M,K] @ bf16 B[K,N] + bias  (f32 out)
// Block tile 128x128, 256 thr (8 waves), wave tile 64x32 (4x2 WMMA accums).
// A tiles: double-buffered TDM (tensor_load_to_lds) with LDS row padding,
// pipelined one tile ahead (wave 0 issues, s_wait_tensorcnt, block barrier).
// B tiles: VMEM b128 loads + transposed 16-bit LDS scatter (K-contiguous
// columns so B fragments are two ds_load_b128 per lane).
// M,N multiples of 128; K multiple of 32.
// ---------------------------------------------------------------------------
__global__ __launch_bounds__(256) void gemm_bf16_wmma(
    const unsigned short* __restrict__ A,
    const unsigned short* __restrict__ B,
    const float* __restrict__ bias,
    float* __restrict__ C,
    int M, int N, int K)
{
    __shared__ unsigned short As[2][128][56]; // [buf][m][k], row 112B: 32 data
                                              // shorts + 24 pad (TDM pad 12 DW)
    __shared__ unsigned short Bs[128][56];    // [n][k] transposed tile

    const int tid     = threadIdx.x;
    const int lane    = tid & 31;
    const int w       = tid >> 5;
    const int wm      = (w >> 2) * 64;       // wave row offset
    const int wn      = (w & 3) * 32;        // wave col offset
    const int bn0     = blockIdx.x * 128;
    const int bm0     = blockIdx.y * 128;

    const int nlane   = lane & 15;
    const int halfsel = lane >> 4;           // 0: lanes 0-15, 1: lanes 16-31
    const int kbA     = halfsel * 16;        // A-frag K base
    const int kbB     = halfsel * 16;        // B-frag K base

    v8f acc[4][2];
#pragma unroll
    for (int i = 0; i < 4; ++i)
#pragma unroll
        for (int j = 0; j < 2; ++j) acc[i][j] = v8f_zero();

    const int brow = tid >> 3;               // 0..31 (k local)
    const int bseg = (tid & 7) * 16;         // n seg

    const unsigned asLds0 = (unsigned)(uintptr_t)&As[0][0][0];
    const unsigned asLds1 = (unsigned)(uintptr_t)&As[1][0][0];

    // Pre-issue TDM for the first A tile (EXEC-independent, once per wave ->
    // restrict to wave 0).
    if (w == 0) {
        tdm_load_2d_pad_bf16(asLds0, A + (size_t)bm0 * K, K, M, 32, 128, K,
                             /*interval 16 DW*/ 3, /*pad 12 DW*/ 11);
    }

    int buf = 0;
    for (int kk = 0; kk < K; kk += 32) {
        // Stage B tile transposed: coalesced b128 read, scattered 16b LDS write
        {
            Frag16 t;
            const u32x4* src = (const u32x4*)(B + (size_t)(kk + brow) * N + bn0 + bseg);
            t.q[0] = src[0];
            t.q[1] = src[1];
#pragma unroll
            for (int i = 0; i < 16; ++i) Bs[bseg + i][brow] = t.s[i];
        }
        if (kk + 32 < K)
            __builtin_prefetch(B + (size_t)(kk + 32 + brow) * N + bn0 + bseg, 0, 0);

        // Pipeline: kick off the next A tile into the other buffer, then wait
        // for the current one (<=1 outstanding keeps the new DMA in flight).
        if (w == 0) {
            if (kk + 32 < K) {
                tdm_load_2d_pad_bf16(buf ? asLds0 : asLds1,
                                     A + (size_t)bm0 * K + kk + 32,
                                     K, M, 32, 128, K, 3, 11);
                __builtin_amdgcn_s_wait_tensorcnt(1);
            } else {
                __builtin_amdgcn_s_wait_tensorcnt(0);
            }
        }
        __syncthreads();

        // A fragments: lane holds row m = nlane, K = kbA..kbA+15 (contiguous)
        v16bf af[4];
#pragma unroll
        for (int i = 0; i < 4; ++i) {
            const unsigned short* r = &As[buf][wm + i * 16 + nlane][0];
            af[i] = load_frag16(r + kbA, r + kbA + 8);
        }
        // B fragments: lane holds col n = nlane, K = kbB..kbB+15 (contiguous)
        v16bf bf[2];
#pragma unroll
        for (int j = 0; j < 2; ++j) {
            const unsigned short* r = &Bs[wn + j * 16 + nlane][0];
            bf[j] = load_frag16(r + kbB, r + kbB + 8);
        }
#pragma unroll
        for (int i = 0; i < 4; ++i)
#pragma unroll
            for (int j = 0; j < 2; ++j)
                acc[i][j] = __builtin_amdgcn_wmma_f32_16x16x32_bf16(
                    false, af[i], false, bf[j], (short)0, acc[i][j], false, false);
        __syncthreads();
        buf ^= 1;
    }

    // Epilogue: C layout row = vgpr r + 8*halfsel, col = nlane
#pragma unroll
    for (int j = 0; j < 2; ++j) {
        const int col = bn0 + wn + j * 16 + nlane;
        const float bv = bias[col];
#pragma unroll
        for (int i = 0; i < 4; ++i) {
#pragma unroll
            for (int r = 0; r < 8; ++r) {
                const int row = bm0 + wm + i * 16 + halfsel * 8 + r;
                C[(size_t)row * N + col] = acc[i][j][r] + bv;
            }
        }
    }
}

// ---------------------------------------------------------------------------
// Kernel: 2D vision RoPE + pack to bf16 head-major (pad head dim 80 -> 96),
// V stored transposed [b][h][96][s] so PV B-fragments are contiguous.
// One thread per (b, token, head).
// ---------------------------------------------------------------------------
__global__ __launch_bounds__(256) void rope_pack(
    const float* __restrict__ qkv,          // [b*s][3840]
    unsigned short* __restrict__ qh,        // [b][h][s][96]
    unsigned short* __restrict__ kh,        // [b][h][s][96]
    unsigned short* __restrict__ vt,        // [b][h][96][s]
    const int* __restrict__ gwp)
{
    const int S = 2304;
    int idx = blockIdx.x * 256 + threadIdx.x;
    if (idx >= 2 * S * 16) return;
    const int h   = idx & 15;
    const int tok = (idx >> 4) % S;
    const int b   = idx / (S * 16);
    const int gw  = gwp[0];

    const float* base = qkv + (size_t)(b * S + tok) * 3840;
    const float* q = base + h * 80;
    const float* k = base + 1280 + h * 80;
    const float* v = base + 2560 + h * 80;

    const float y = (float)(tok / gw);
    const float x = (float)(tok % gw);
    const float qscale = 0.11180339887498949f;   // 80^-0.5

    const size_t off = (size_t)(b * 16 + h) * S * 96 + (size_t)tok * 96;
    unsigned short* qo = qh + off;
    unsigned short* ko = kh + off;

#pragma unroll 4
    for (int p = 0; p < 40; ++p) {
        const float inv = __expf(-(float)(p % 20) * (9.210340371976184f / 20.0f));
        const float pos = (p < 20) ? y : x;
        float sn, cs;
        __sincosf(pos * inv, &sn, &cs);
        const float qe = q[2 * p], qodd = q[2 * p + 1];
        const float ke = k[2 * p], kodd = k[2 * p + 1];
        qo[2 * p]     = f2bf((qe * cs - qodd * sn) * qscale);
        qo[2 * p + 1] = f2bf((qodd * cs + qe * sn) * qscale);
        ko[2 * p]     = f2bf(ke * cs - kodd * sn);
        ko[2 * p + 1] = f2bf(kodd * cs + ke * sn);
    }
#pragma unroll
    for (int c = 80; c < 96; ++c) { qo[c] = 0; ko[c] = 0; }   // zero pad (read by QK^T)

    const size_t vbase = (size_t)(b * 16 + h) * 96 * (size_t)S + tok;
#pragma unroll 8
    for (int c = 0; c < 80; ++c) vt[vbase + (size_t)c * S] = f2bf(v[c]);
}

// ---------------------------------------------------------------------------
// Kernel: flash attention. 1 wave per 16-row q tile, 32 keys per iteration.
// S tile = QK^T (6 wmma), online softmax via half-wave shfl reductions,
// P re-swizzled C-layout -> A-layout through per-wave LDS, PV = 5 wmma.
// ---------------------------------------------------------------------------
__global__ __launch_bounds__(128) void flash_attn_wmma(
    const unsigned short* __restrict__ qh,
    const unsigned short* __restrict__ kh,
    const unsigned short* __restrict__ vt,
    unsigned short* __restrict__ enc)       // [b][s][1280] bf16
{
    const int S = 2304;
    __shared__ unsigned short Pls[4][16][56];   // per-wave P patch

    const int lane    = threadIdx.x & 31;
    const int w       = threadIdx.x >> 5;
    const int h       = blockIdx.y;
    const int b       = blockIdx.z;
    const int q0      = (blockIdx.x * 4 + w) * 16;

    const int nlane   = lane & 15;
    const int halfsel = lane >> 4;
    const int kbA     = halfsel * 16;
    const int kbB     = halfsel * 16;

    const size_t headQK = (size_t)(b * 16 + h) * S * 96;
    const size_t headV  = (size_t)(b * 16 + h) * 96 * (size_t)S;

    // Q fragments for the 3 K-chunks of the padded 96 head dim
    v16bf qa[3];
#pragma unroll
    for (int c = 0; c < 3; ++c) {
        const unsigned short* p = qh + headQK + (size_t)(q0 + nlane) * 96 + c * 32;
        qa[c] = load_frag16(p + kbA, p + kbA + 8);
    }

    v8f o[5];
#pragma unroll
    for (int j = 0; j < 5; ++j) o[j] = v8f_zero();
    float mr[8], lr[8];
#pragma unroll
    for (int r = 0; r < 8; ++r) { mr[r] = -1e30f; lr[r] = 0.f; }

    for (int k0 = 0; k0 < S; k0 += 32) {
        // ---- S = Q @ K^T : two 16x16 column tiles, 3 K-chunks each ----
        v8f s0 = v8f_zero(), s1 = v8f_zero();
#pragma unroll
        for (int c = 0; c < 3; ++c) {
            const unsigned short* pk0 = kh + headQK + (size_t)(k0 + nlane) * 96 + c * 32 + kbB;
            const unsigned short* pk1 = kh + headQK + (size_t)(k0 + 16 + nlane) * 96 + c * 32 + kbB;
            v16bf b0 = load_frag16(pk0, pk0 + 8);
            v16bf b1 = load_frag16(pk1, pk1 + 8);
            s0 = __builtin_amdgcn_wmma_f32_16x16x32_bf16(false, qa[c], false, b0,
                                                         (short)0, s0, false, false);
            s1 = __builtin_amdgcn_wmma_f32_16x16x32_bf16(false, qa[c], false, b1,
                                                         (short)0, s1, false, false);
        }
        // ---- online softmax: rows live across 16-lane halves ----
        float p0[8], p1[8];
#pragma unroll
        for (int r = 0; r < 8; ++r) {
            float v0 = s0[r], v1 = s1[r];
            float mx = fmaxf(v0, v1);
            mx = fmaxf(mx, __shfl_xor(mx, 1, 32));
            mx = fmaxf(mx, __shfl_xor(mx, 2, 32));
            mx = fmaxf(mx, __shfl_xor(mx, 4, 32));
            mx = fmaxf(mx, __shfl_xor(mx, 8, 32));
            const float mnew = fmaxf(mr[r], mx);
            const float corr = __expf(mr[r] - mnew);
            const float e0 = __expf(v0 - mnew);
            const float e1 = __expf(v1 - mnew);
            float rs = e0 + e1;
            rs += __shfl_xor(rs, 1, 32);
            rs += __shfl_xor(rs, 2, 32);
            rs += __shfl_xor(rs, 4, 32);
            rs += __shfl_xor(rs, 8, 32);
            lr[r] = lr[r] * corr + rs;
            mr[r] = mnew;
#pragma unroll
            for (int j = 0; j < 5; ++j) o[j][r] *= corr;
            p0[r] = e0;
            p1[r] = e1;
        }
        // ---- re-swizzle P (C-layout) -> A-layout through LDS ----
#pragma unroll
        for (int r = 0; r < 8; ++r) {
            Pls[w][halfsel * 8 + r][nlane]      = f2bf(p0[r]);
            Pls[w][halfsel * 8 + r][nlane + 16] = f2bf(p1[r]);
        }
        __syncthreads();
        const unsigned short* pp = &Pls[w][nlane][0];
        const v16bf pa = load_frag16(pp + kbA, pp + kbA + 8);
        // ---- O += P @ V  (V transposed: contiguous B fragments) ----
#pragma unroll
        for (int j = 0; j < 5; ++j) {
            const unsigned short* pv = vt + headV + (size_t)(j * 16 + nlane) * S + k0 + kbB;
            v16bf bv = load_frag16(pv, pv + 8);
            o[j] = __builtin_amdgcn_wmma_f32_16x16x32_bf16(false, pa, false, bv,
                                                           (short)0, o[j], false, false);
        }
        __syncthreads();
    }

    // ---- normalize and scatter to enc[b][s][h*80 + col] ----
#pragma unroll
    for (int r = 0; r < 8; ++r) {
        const float inv = 1.0f / lr[r];
        const int tok = q0 + halfsel * 8 + r;
#pragma unroll
        for (int j = 0; j < 5; ++j) {
            const int col = j * 16 + nlane;
            enc[(size_t)(b * S + tok) * 1280 + h * 80 + col] = f2bf(o[j][r] * inv);
        }
    }
}

// ---------------------------------------------------------------------------
// Host launcher
// ---------------------------------------------------------------------------
extern "C" void kernel_launch(void* const* d_in, const int* in_sizes, int n_in,
                              void* d_out, int out_size, void* d_ws, size_t ws_size,
                              hipStream_t stream) {
    (void)in_sizes; (void)n_in; (void)out_size; (void)ws_size;

    const float* x      = (const float*)d_in[0];   // [2,2304,1280]
    const float* qkv_w  = (const float*)d_in[1];   // [1280,3840]
    const float* qkv_b  = (const float*)d_in[2];   // [3840]
    const float* proj_w = (const float*)d_in[3];   // [1280,1280]
    const float* proj_b = (const float*)d_in[4];   // [1280]
    const int*   gw_p   = (const int*)d_in[6];     // grid_w (48)
    float* out = (float*)d_out;                    // [2,2304,1280]

    const int B = 2, S = 2304, D = 1280;
    const int M = B * S;                            // 4608

    // Workspace carve-up (all offsets 256B aligned)
    char* ws = (char*)d_ws;
    size_t off = 0;
    unsigned short* xb     = (unsigned short*)(ws + off); off += (size_t)M * D * 2;        // 11,796,480
    unsigned short* wqkvb  = (unsigned short*)(ws + off); off += (size_t)D * 3840 * 2;     //  9,830,400
    unsigned short* wprojb = (unsigned short*)(ws + off); off += (size_t)D * D * 2;        //  3,276,800
    float*          qkvf   = (float*)(ws + off);          off += (size_t)M * 3840 * 4;     // 70,778,880
    unsigned short* qh     = (unsigned short*)(ws + off); off += (size_t)B * 16 * S * 96 * 2;
    unsigned short* kh     = (unsigned short*)(ws + off); off += (size_t)B * 16 * S * 96 * 2;
    unsigned short* vt     = (unsigned short*)(ws + off); off += (size_t)B * 16 * S * 96 * 2;
    unsigned short* encb   = (unsigned short*)(ws + off); off += (size_t)M * D * 2;

    // 1) convert inputs to bf16
    const int nx = M * D;            // 5,898,240
    const int nw1 = D * 3840;        // 4,915,200
    const int nw2 = D * D;           // 1,638,400
    cvt_f32_bf16<<<(nx  + 255) / 256, 256, 0, stream>>>(x,      xb,     nx);
    cvt_f32_bf16<<<(nw1 + 255) / 256, 256, 0, stream>>>(qkv_w,  wqkvb,  nw1);
    cvt_f32_bf16<<<(nw2 + 255) / 256, 256, 0, stream>>>(proj_w, wprojb, nw2);

    // 2) QKV = x @ qkv_w + qkv_b      (4608 x 3840 x 1280)
    gemm_bf16_wmma<<<dim3(3840 / 128, M / 128), 256, 0, stream>>>(
        xb, wqkvb, qkv_b, qkvf, M, 3840, D);

    // 3) RoPE + head-major bf16 pack (+ V transpose)
    rope_pack<<<(B * S * 16) / 256, 256, 0, stream>>>(qkvf, qh, kh, vt, gw_p);

    // 4) flash attention -> enc (bf16)
    flash_attn_wmma<<<dim3(S / 64, 16, B), 128, 0, stream>>>(qh, kh, vt, encb);

    // 5) out = enc @ proj_w + proj_b  (4608 x 1280 x 1280)
    gemm_bf16_wmma<<<dim3(D / 128, M / 128), 256, 0, stream>>>(
        encb, wprojb, proj_b, out, M, D, D);
}
